// ContextEncoder_11862699671826
// MI455X (gfx1250) — matrix operand order repaired and checked by reading, not compiled, not run on hardware
//
#include <hip/hip_runtime.h>
#include <math.h>

typedef __attribute__((ext_vector_type(16))) __bf16 v16bf;
typedef __attribute__((ext_vector_type(8)))  float v8f;
typedef __attribute__((ext_vector_type(4)))  unsigned int v4u;

union BF16Frag { v16bf v; v4u q[2]; unsigned short u[16]; };

// ---------------- constants ----------------
static constexpr int Bn = 512, Ln = 10, Hn = 256, Dn = 512;
static constexpr int NROWS = Bn * Ln * Ln;          // 51200
static constexpr float NTOT = 1310720.0f;           // B*L*H
static constexpr float BN_EPS = 1e-5f;
static constexpr float Ecst  = 0.01f;

// workspace layout (float offsets)
static constexpr size_t OFF_PA   = 0;
static constexpr size_t OFF_QA   = 1310720;
static constexpr size_t OFF_PS   = 2621440;
static constexpr size_t OFF_QS   = 3932160;
static constexpr size_t OFF_SQA  = 5242880;
static constexpr size_t OFF_SQS  = 5373952;
static constexpr size_t OFF_STAT = 5505024;   // 4 qstats + 40 layer2 stats (zeroed each call)
static constexpr size_t OFF_BN1  = 5505072;   // 40
static constexpr size_t OFF_BN2  = 5505112;   // 40
static constexpr size_t OFF_WIJ  = 5505152;   // 51200
static constexpr size_t OFF_H2A  = 5556352;   // bf16 [51200,256] -> 6553600 float slots
static constexpr size_t OFF_H2S  = 12109952;
static constexpr size_t OFF_UTTB = 18663552;  // bf16 utt   [5120,512]  -> 1310720 float slots
static constexpr size_t OFF_W1A  = 19974272;  // bf16 a_w1  [256,1024]  ->  131072
static constexpr size_t OFF_W1S  = 20105344;  // bf16 s_w1
static constexpr size_t OFF_W2A  = 20236416;  // bf16 a_w2  [256,256]   ->   32768
static constexpr size_t OFF_W2S  = 20269184;  // bf16 s_w2
// total ~81 MiB

// ---------------- helpers ----------------
__device__ __forceinline__ unsigned short f2bf(float f) {
    __bf16 h = (__bf16)f;                        // native v_cvt on gfx1250
    return __builtin_bit_cast(unsigned short, h);
}
__device__ __forceinline__ float bf2f(unsigned short h) {
    return __uint_as_float(((unsigned)h) << 16);
}
__device__ __forceinline__ float waveReduce(float v) {
    #pragma unroll
    for (int m = 16; m > 0; m >>= 1) v += __shfl_xor(v, m, 32);
    return v;
}
__device__ __forceinline__ float softplusf(float x) {
    return (x > 20.0f) ? x : log1pf(expf(x));
}

// ---------------- kernel 0: f32 -> bf16 pre-conversion ----------------
__global__ void cvt_bf16_kernel(const float* __restrict__ src,
                                unsigned short* __restrict__ dst, int n) {
    const int i = blockIdx.x * 256 + threadIdx.x;
    if (i < n) dst[i] = f2bf(src[i]);
}

// ---------------- kernel 1: P/Q GEMMs (bf16 WMMA, bf16 operands) ----------------
// C[5120,256] = A[5120,512]bf16 * Wbf16[:, kOff:kOff+512]^T (+bias). W row-major [256, ldw].
__global__ void gemm1_kernel(const unsigned short* __restrict__ A,
                             const unsigned short* __restrict__ W,
                             int ldw, int kOff, const float* __restrict__ bias,
                             float* __restrict__ C) {
    const int tid  = threadIdx.x;
    const int lane = tid & 31;
    const int wave = blockIdx.x * 8 + (tid >> 5);
    const int mTile = wave >> 2;          // 0..319
    const int nTile = wave & 3;           // 64-col slab
    const int lm = lane & 15, grp = lane >> 4;
    const unsigned short* Arow = A + (size_t)(mTile * 16 + lm) * Dn;

    v8f acc[4] = {};
    for (int kt = 0; kt < 16; ++kt) {
        BF16Frag af;
        const unsigned short* ap = Arow + kt * 32 + 8 * grp;
        af.q[0] = *(const v4u*)(ap);          // k run [kb, kb+8)
        af.q[1] = *(const v4u*)(ap + 16);     // k run [kb+16, kb+24)
        const int kbB = kOff + kt * 32 + 8 * grp;
        #pragma unroll
        for (int sub = 0; sub < 4; ++sub) {
            const int n = nTile * 64 + sub * 16 + lm;
            const unsigned short* bp = W + (size_t)n * ldw + kbB;
            BF16Frag bf;
            bf.q[0] = *(const v4u*)(bp);
            bf.q[1] = *(const v4u*)(bp + 16);
            acc[sub] = __builtin_amdgcn_wmma_f32_16x16x32_bf16(
                false, af.v, false, bf.v, (short)0, acc[sub], false, false);
        }
    }
    #pragma unroll
    for (int sub = 0; sub < 4; ++sub) {
        const int c  = nTile * 64 + sub * 16 + lm;
        const float bv = bias ? bias[c] : 0.0f;
        #pragma unroll
        for (int v = 0; v < 8; ++v) {
            const int row = mTile * 16 + v + 8 * grp;
            C[(size_t)row * Hn + c] = acc[sub][v] + bv;
        }
    }
}

// ---------------- kernel 2: Q stats + SQ[b,h] ----------------
__global__ void qstats_kernel(const float* __restrict__ QA, const float* __restrict__ QS,
                              float* __restrict__ SQA, float* __restrict__ SQS,
                              float* __restrict__ qstat) {
    __shared__ float red[256];
    const int b = blockIdx.x, h = threadIdx.x;
    const float* Qs[2] = {QA, QS};
    float* SQs[2] = {SQA, SQS};
    for (int br = 0; br < 2; ++br) {
        float s = 0.f, s2 = 0.f;
        #pragma unroll
        for (int j = 0; j < Ln; ++j) {
            float q = Qs[br][((size_t)b * Ln + j) * Hn + h];
            s += q; s2 += q * q;
        }
        SQs[br][(size_t)b * Hn + h] = s;
        red[h] = s; __syncthreads();
        for (int st = 128; st > 0; st >>= 1) { if (h < st) red[h] += red[h + st]; __syncthreads(); }
        if (h == 0) atomicAdd(&qstat[br * 2 + 0], red[0]);
        __syncthreads();
        red[h] = s2; __syncthreads();
        for (int st = 128; st > 0; st >>= 1) { if (h < st) red[h] += red[h + st]; __syncthreads(); }
        if (h == 0) atomicAdd(&qstat[br * 2 + 1], red[0]);
        __syncthreads();
    }
}

// ---------------- kernel 3: BN1 scale/shift per (branch, i) ----------------
__global__ void bn1_kernel(const float* __restrict__ PA, const float* __restrict__ SQA,
                           const float* __restrict__ PS, const float* __restrict__ SQS,
                           const float* __restrict__ qstat,
                           const float* __restrict__ a_g1, const float* __restrict__ a_be1,
                           const float* __restrict__ s_g1, const float* __restrict__ s_be1,
                           float* __restrict__ bn1) {
    __shared__ float red[256];
    const int blk = blockIdx.x;            // 0..19
    const int branch = blk / Ln, i = blk % Ln;
    const float* P  = branch ? PS  : PA;
    const float* SQ = branch ? SQS : SQA;
    const float* g  = branch ? s_g1  : a_g1;
    const float* be = branch ? s_be1 : a_be1;
    const float qs  = qstat[branch * 2 + 0];
    const float qs2 = qstat[branch * 2 + 1];
    const int tid = threadIdx.x;
    float sp = 0.f, sp2 = 0.f, spq = 0.f;
    for (int idx = tid; idx < Bn * Hn; idx += 256) {
        const int b = idx >> 8, h = idx & 255;
        const float pv = P[((size_t)b * Ln + i) * Hn + h];
        const float sq = SQ[idx];
        sp += pv; sp2 += pv * pv; spq += pv * sq;
    }
    float vals[3] = {sp, sp2, spq}, out[3];
    for (int s = 0; s < 3; ++s) {
        red[tid] = vals[s]; __syncthreads();
        for (int st = 128; st > 0; st >>= 1) { if (tid < st) red[tid] += red[tid + st]; __syncthreads(); }
        if (tid == 0) out[s] = red[0];
        __syncthreads();
    }
    if (tid == 0) {
        const float mean = (10.0f * out[0] + qs) / NTOT;
        const float e2   = (10.0f * out[1] + 2.0f * out[2] + qs2) / NTOT;
        const float var  = e2 - mean * mean;
        const float sc   = g[i] * rsqrtf(var + BN_EPS);
        bn1[branch * 20 + i]      = sc;
        bn1[branch * 20 + 10 + i] = be[i] - mean * sc;
    }
}

// ---------------- kernel 4: layer-2 GEMM, fused bn1+act, bn2 stat partials ----------------
__global__ void layer2_kernel(const float* __restrict__ P, const float* __restrict__ Q,
                              const float* __restrict__ bn1scale, const float* __restrict__ bn1shift,
                              const unsigned short* __restrict__ W2, const float* __restrict__ b2,
                              unsigned short* __restrict__ H2,
                              float* __restrict__ statSum, float* __restrict__ statSsq,
                              int actTanh) {
    __shared__ unsigned short h1[64 * 264];
    __shared__ float rsum[64], rssq[64];
    const int tid = threadIdx.x;
    const int Rbase = blockIdx.x * 64;
    if (tid < 64) { rsum[tid] = 0.f; rssq[tid] = 0.f; }
    // phase 1: activation tile (bf16) into LDS
    for (int row = 0; row < 64; ++row) {
        const int R = Rbase + row;
        const int b = R / 100, rem = R - b * 100;
        const int i = rem / 10, j = rem - i * 10;
        const float p = P[((size_t)b * Ln + i) * Hn + tid];
        const float q = Q[((size_t)b * Ln + j) * Hn + tid];
        float y = bn1scale[i] * (p + q) + bn1shift[i];
        y = actTanh ? tanhf(y) : fmaxf(y, 0.0f);
        h1[row * 264 + tid] = f2bf(y);
    }
    __syncthreads();
    // phase 2: WMMA — wave covers 16 rows x 128 cols
    const int lane = tid & 31, wv = tid >> 5;
    const int rowSub = wv & 3, colHalf = wv >> 2;
    const int lm = lane & 15, grp = lane >> 4;
    v8f acc[8] = {};
    for (int kt = 0; kt < 8; ++kt) {
        BF16Frag af;
        const unsigned short* ap = &h1[(rowSub * 16 + lm) * 264 + kt * 32 + 8 * grp];
        af.q[0] = *(const v4u*)(ap);
        af.q[1] = *(const v4u*)(ap + 16);
        const int kb = kt * 32 + 8 * grp;
        #pragma unroll
        for (int ct = 0; ct < 8; ++ct) {
            const int n = colHalf * 128 + ct * 16 + lm;
            const unsigned short* bp = W2 + (size_t)n * Hn + kb;
            BF16Frag bf;
            bf.q[0] = *(const v4u*)(bp);
            bf.q[1] = *(const v4u*)(bp + 16);
            acc[ct] = __builtin_amdgcn_wmma_f32_16x16x32_bf16(
                false, af.v, false, bf.v, (short)0, acc[ct], false, false);
        }
    }
    // epilogue: add bias, store bf16, accumulate per-row stats
    #pragma unroll
    for (int ct = 0; ct < 8; ++ct) {
        const int c = colHalf * 128 + ct * 16 + lm;
        const float bv = b2[c];
        #pragma unroll
        for (int v = 0; v < 8; ++v) {
            const int rl = rowSub * 16 + v + 8 * grp;
            const float val = acc[ct][v] + bv;
            acc[ct][v] = val;
            H2[(size_t)(Rbase + rl) * Hn + c] = f2bf(val);
        }
    }
    #pragma unroll
    for (int v = 0; v < 8; ++v) {
        float s = 0.f, s2 = 0.f;
        #pragma unroll
        for (int ct = 0; ct < 8; ++ct) { const float x = acc[ct][v]; s += x; s2 += x * x; }
        const int rl = rowSub * 16 + v + 8 * grp;
        atomicAdd(&rsum[rl], s);
        atomicAdd(&rssq[rl], s2);
    }
    __syncthreads();
    if (tid < 64) {
        const int R = Rbase + tid;
        const int b = R / 100, rem = R - b * 100, i = rem / 10;
        atomicAdd(&statSum[i], rsum[tid]);
        atomicAdd(&statSsq[i], rssq[tid]);
    }
}

// ---------------- kernel 5: finalize BN2 ----------------
__global__ void bn2_kernel(const float* __restrict__ stat2,
                           const float* __restrict__ a_g2, const float* __restrict__ a_be2,
                           const float* __restrict__ s_g2, const float* __restrict__ s_be2,
                           float* __restrict__ bn2) {
    const int t = threadIdx.x;
    if (t < 20) {
        const int branch = t / 10, i = t % 10;
        const float mean = stat2[branch * 20 + i] / NTOT;
        const float e2   = stat2[branch * 20 + 10 + i] / NTOT;
        const float var  = e2 - mean * mean;
        const float* g  = branch ? s_g2  : a_g2;
        const float* be = branch ? s_be2 : a_be2;
        const float sc = g[i] * rsqrtf(var + BN_EPS);
        bn2[branch * 20 + i]      = sc;
        bn2[branch * 20 + 10 + i] = be[i] - mean * sc;
    }
}

// ---------------- kernel 6: heads + elementwise -> wij ----------------
__global__ void heads_kernel(const unsigned short* __restrict__ H2A,
                             const unsigned short* __restrict__ H2S,
                             const float* __restrict__ bn2,
                             const float* __restrict__ a_w3, const float* __restrict__ a_b3,
                             const float* __restrict__ mu_w, const float* __restrict__ mu_b,
                             const float* __restrict__ sg_w, const float* __restrict__ sg_b,
                             const float* __restrict__ aij_eps, const float* __restrict__ sij_eps,
                             float* __restrict__ wij) {
    const int tid = threadIdx.x, lane = tid & 31, wv = tid >> 5;
    const int R = blockIdx.x * 8 + wv;
    const int b = R / 100, rem = R - b * 100, i = rem / 10;
    const float sa = bn2[i],      ta = bn2[10 + i];
    const float ss = bn2[20 + i], ts = bn2[30 + i];
    float da = 0.f, dm = 0.f, dg = 0.f;
    #pragma unroll
    for (int e0 = 0; e0 < 8; ++e0) {
        const int e = lane * 8 + e0;
        const float va = fmaxf(sa * bf2f(H2A[(size_t)R * Hn + e]) + ta, 0.0f);
        da += va * a_w3[e];
        const float vs = tanhf(ss * bf2f(H2S[(size_t)R * Hn + e]) + ts);
        dm += vs * mu_w[e];
        dg += vs * sg_w[e];
    }
    da = waveReduce(da); dm = waveReduce(dm); dg = waveReduce(dg);
    if (lane == 0) {
        const float aij_mij   = da + a_b3[0];
        const float sij_mu    = dm + mu_b[0];
        const float sij_sigma = dg + sg_b[0];
        const float aij_mu  = softplusf(aij_mij) + Ecst;
        const float aij_std = sqrtf(softplusf((1.0f - aij_mu) * aij_mu) + Ecst);
        const float aij = softplusf(aij_eps[R] * aij_std + aij_mu) + Ecst;
        const float sij = sij_eps[R] * sqrtf(aij * sij_sigma * sij_sigma) + aij * sij_mu;
        wij[R] = aij * sij;
    }
}

// ---------------- kernel 7: per-dialog softmax, message, GRU step ----------------
__global__ void agg_gru_kernel(const float* __restrict__ utt, const int* __restrict__ clens,
                               const float* __restrict__ wij,
                               const float* __restrict__ wi, const float* __restrict__ wh,
                               const float* __restrict__ bi, const float* __restrict__ bh,
                               float* __restrict__ out) {
    __shared__ float p[16];
    __shared__ float msg[512];
    __shared__ float h0s[256];
    __shared__ float gi[768], gh[768];
    const int b = blockIdx.x, tid = threadIdx.x;
    const int anchor = clens[b] - 1;
    if (tid == 0) {
        float mx = -1e30f;
        for (int j = 0; j < anchor; ++j) mx = fmaxf(mx, wij[b * 100 + anchor * 10 + j]);
        float ex[10], ssum = 0.f;
        for (int j = 0; j < anchor; ++j) { ex[j] = expf(wij[b * 100 + anchor * 10 + j] - mx); ssum += ex[j]; }
        for (int j = 0; j < 10; ++j) p[j] = (j < anchor) ? ex[j] / ssum : 0.0f;
    }
    __syncthreads();
    for (int d = tid; d < Dn; d += 256) {
        float m = 0.f;
        #pragma unroll
        for (int j = 0; j < Ln; ++j) m += p[j] * utt[((size_t)b * Ln + j) * Dn + d];
        msg[d] = m;
    }
    if (tid < Hn) {
        const float* ue = utt + ((size_t)b * Ln + anchor) * Dn;
        h0s[tid] = ue[tid] + ue[Hn + tid];
    }
    __syncthreads();
    for (int g = tid; g < 768; g += 256) {
        float s = bi[g];
        const float* wr = wi + (size_t)g * Dn;
        for (int d = 0; d < Dn; ++d) s += msg[d] * wr[d];
        gi[g] = s;
        float s2 = bh[g];
        const float* whr = wh + (size_t)g * Hn;
        for (int h = 0; h < Hn; ++h) s2 += h0s[h] * whr[h];
        gh[g] = s2;
    }
    __syncthreads();
    if (tid < Hn) {
        const float r = 1.0f / (1.0f + expf(-(gi[tid] + gh[tid])));
        const float z = 1.0f / (1.0f + expf(-(gi[256 + tid] + gh[256 + tid])));
        const float n = tanhf(gi[512 + tid] + r * gh[512 + tid]);
        out[(size_t)b * Hn + tid] = (1.0f - z) * n + z * h0s[tid];
    }
}

// ---------------- host launcher ----------------
extern "C" void kernel_launch(void* const* d_in, const int* in_sizes, int n_in,
                              void* d_out, int out_size, void* d_ws, size_t ws_size,
                              hipStream_t stream) {
    const float* utt      = (const float*)d_in[0];
    const float* aij_eps  = (const float*)d_in[1];
    const float* sij_eps  = (const float*)d_in[2];
    const int*   clens    = (const int*)d_in[3];
    const float* a_w1 = (const float*)d_in[4];  const float* a_b1 = (const float*)d_in[5];
    const float* a_g1 = (const float*)d_in[6];  const float* a_be1= (const float*)d_in[7];
    const float* a_w2 = (const float*)d_in[8];  const float* a_b2 = (const float*)d_in[9];
    const float* a_g2 = (const float*)d_in[10]; const float* a_be2= (const float*)d_in[11];
    const float* a_w3 = (const float*)d_in[12]; const float* a_b3 = (const float*)d_in[13];
    const float* s_w1 = (const float*)d_in[14]; const float* s_b1 = (const float*)d_in[15];
    const float* s_g1 = (const float*)d_in[16]; const float* s_be1= (const float*)d_in[17];
    const float* s_w2 = (const float*)d_in[18]; const float* s_b2 = (const float*)d_in[19];
    const float* s_g2 = (const float*)d_in[20]; const float* s_be2= (const float*)d_in[21];
    const float* mu_w = (const float*)d_in[22]; const float* mu_b = (const float*)d_in[23];
    const float* sg_w = (const float*)d_in[24]; const float* sg_b = (const float*)d_in[25];
    const float* gru_wi = (const float*)d_in[26]; const float* gru_wh = (const float*)d_in[27];
    const float* gru_bi = (const float*)d_in[28]; const float* gru_bh = (const float*)d_in[29];

    float* ws = (float*)d_ws;
    float* PA = ws + OFF_PA;  float* QA = ws + OFF_QA;
    float* PS = ws + OFF_PS;  float* QS = ws + OFF_QS;
    float* SQA = ws + OFF_SQA; float* SQS = ws + OFF_SQS;
    float* qstat = ws + OFF_STAT;          // 4 floats
    float* stat2 = qstat + 4;              // 40 floats
    float* bn1 = ws + OFF_BN1;
    float* bn2 = ws + OFF_BN2;
    float* wij = ws + OFF_WIJ;
    unsigned short* H2A  = (unsigned short*)(ws + OFF_H2A);
    unsigned short* H2S  = (unsigned short*)(ws + OFF_H2S);
    unsigned short* UTTB = (unsigned short*)(ws + OFF_UTTB);
    unsigned short* W1A  = (unsigned short*)(ws + OFF_W1A);
    unsigned short* W1S  = (unsigned short*)(ws + OFF_W1S);
    unsigned short* W2A  = (unsigned short*)(ws + OFF_W2A);
    unsigned short* W2S  = (unsigned short*)(ws + OFF_W2S);

    // zero atomically-accumulated stats (graph-capture-safe)
    hipMemsetAsync(qstat, 0, 48 * sizeof(float), stream);

    // pre-convert GEMM operands to bf16 (bandwidth-trivial)
    cvt_bf16_kernel<<<(2621440 + 255) / 256, 256, 0, stream>>>(utt,  UTTB, 2621440);
    cvt_bf16_kernel<<<(262144  + 255) / 256, 256, 0, stream>>>(a_w1, W1A,  262144);
    cvt_bf16_kernel<<<(262144  + 255) / 256, 256, 0, stream>>>(s_w1, W1S,  262144);
    cvt_bf16_kernel<<<(65536   + 255) / 256, 256, 0, stream>>>(a_w2, W2A,  65536);
    cvt_bf16_kernel<<<(65536   + 255) / 256, 256, 0, stream>>>(s_w2, W2S,  65536);

    // layer 1 factored GEMMs (bias folded into P)
    gemm1_kernel<<<160, 256, 0, stream>>>(UTTB, W1A, 1024, 0,   a_b1,   PA);
    gemm1_kernel<<<160, 256, 0, stream>>>(UTTB, W1A, 1024, 512, nullptr, QA);
    gemm1_kernel<<<160, 256, 0, stream>>>(UTTB, W1S, 1024, 0,   s_b1,   PS);
    gemm1_kernel<<<160, 256, 0, stream>>>(UTTB, W1S, 1024, 512, nullptr, QS);

    qstats_kernel<<<512, 256, 0, stream>>>(QA, QS, SQA, SQS, qstat);
    bn1_kernel<<<20, 256, 0, stream>>>(PA, SQA, PS, SQS, qstat,
                                       a_g1, a_be1, s_g1, s_be1, bn1);

    layer2_kernel<<<800, 256, 0, stream>>>(PA, QA, bn1 + 0,  bn1 + 10, W2A, a_b2,
                                           H2A, stat2 + 0,  stat2 + 10, 0);
    layer2_kernel<<<800, 256, 0, stream>>>(PS, QS, bn1 + 20, bn1 + 30, W2S, s_b2,
                                           H2S, stat2 + 20, stat2 + 30, 1);

    bn2_kernel<<<1, 32, 0, stream>>>(stat2, a_g2, a_be2, s_g2, s_be2, bn2);

    heads_kernel<<<NROWS / 8, 256, 0, stream>>>(H2A, H2S, bn2, a_w3, a_b3,
                                                mu_w, mu_b, sg_w, sg_b,
                                                aij_eps, sij_eps, wij);

    agg_gru_kernel<<<Bn, 256, 0, stream>>>(utt, clens, wij, gru_wi, gru_wh,
                                           gru_bi, gru_bh, (float*)d_out);
}
